// MultiQueryAttention_4071628997210
// MI455X (gfx1250) — matrix operand order repaired and checked by reading, not compiled, not run on hardware
//
#include <hip/hip_runtime.h>
#include <stdint.h>

// ---------------------------------------------------------------------------
// Types for CDNA5 WMMA (wave32): v_wmma_f32_16x16x32_bf16
// ---------------------------------------------------------------------------
typedef __attribute__((ext_vector_type(16))) __bf16 bf16x16;
typedef __attribute__((ext_vector_type(8)))  __bf16 bf16x8;
typedef __attribute__((ext_vector_type(8)))  float  f32x8;

constexpr int B_    = 4;
constexpr int T_    = 2048;
constexpr int HID_  = 2048;
constexpr int NH_   = 16;
constexpr int NKV_  = 2;
constexpr int HD_   = 128;
constexpr int LDQ_  = NH_ * HD_ + 2 * NKV_ * HD_;   // 2560 (q | k | v concatenated)
constexpr int ROWS_ = B_ * T_;                      // 8192 tokens

__device__ __forceinline__ f32x8 wmma_bf16(bf16x16 a, bf16x16 b, f32x8 c) {
  // D = A(16x32) * B(32x16) + C(16x16 f32)
  return __builtin_amdgcn_wmma_f32_16x16x32_bf16(
      /*neg_a=*/false, a, /*neg_b=*/false, b,
      /*c_mod=*/(short)0, c, /*reuse_a=*/false, /*reuse_b=*/false);
}

union bf16x16_u { bf16x16 v; bf16x8 h[2]; };

// A-fragment (16x32, row-major source, leading dim ld).
// Layout: row M=lane&15, elems e -> K = 16*(lane>>4) + e (contiguous, matches B).
__device__ __forceinline__ bf16x16
load_a_frag(const __bf16* A, int ld, int m0, int k0, int lane) {
  const int row = m0 + (lane & 15);
  const int kb  = k0 + ((lane >> 4) << 4);
  const __bf16* p = A + (size_t)row * ld + kb;
  bf16x16_u u;
  u.h[0] = *(const bf16x8*)(p);
  u.h[1] = *(const bf16x8*)(p + 8);
  return u.v;
}

// B-fragment (32x16) sourced from row-major W[N,K] (computing X*W^T):
// lane col n = lane&15, elems e -> K = k0 + 16*(lane>>4) + e  (contiguous 32B).
__device__ __forceinline__ bf16x16
load_b_frag(const __bf16* W, int ld, int n0, int k0, int lane) {
  const int n = n0 + (lane & 15);
  const int k = k0 + ((lane >> 4) << 4);
  return *(const bf16x16*)(W + (size_t)n * ld + k);
}

// CDNA5 async copy: per-lane 16B global -> LDS, tracked by ASYNCcnt.
// GV mode: vaddr = 64-bit per-lane global address, saddr = off (NULL).
// The LDS destination is the low 32 bits of the generic shared pointer.
__device__ __forceinline__ void
async_g2lds_b128(void* lds_dst, const void* gsrc) {
  const uint32_t ldsoff = (uint32_t)(uintptr_t)lds_dst;
  const uint64_t gaddr  = (uint64_t)(uintptr_t)gsrc;
  asm volatile("global_load_async_to_lds_b128 %0, %1, off"
               :: "v"(ldsoff), "v"(gaddr)
               : "memory");
}
__device__ __forceinline__ void wait_asynccnt0() {
  asm volatile("s_wait_asynccnt 0x0" ::: "memory");
}

// ---------------------------------------------------------------------------
// f32 -> bf16 conversion (grid-stride)
// ---------------------------------------------------------------------------
__global__ void cvt_f32_bf16_kernel(const float* __restrict__ in,
                                    __bf16* __restrict__ out, long n) {
  long i = (long)blockIdx.x * blockDim.x + threadIdx.x;
  const long stride = (long)gridDim.x * blockDim.x;
  for (; i < n; i += stride) out[i] = (__bf16)in[i];
}

// ---------------------------------------------------------------------------
// Tiled GEMM: C[M,N] = A[M,K] * W[N,K]^T   (A,W bf16 row-major, C f32 or bf16)
// Block: 256 thr = 8 waves; tile 128x128; wave tile 32x64 (2x4 WMMA accums).
// Whole operand set is L2-resident (192MB), so fragments load straight from
// global; next-k lines are prefetched toward the WGP.
// ---------------------------------------------------------------------------
template <bool OUT_BF16>
__global__ void __launch_bounds__(256)
gemm_bf16_kernel(const __bf16* __restrict__ A, const __bf16* __restrict__ W,
                 void* __restrict__ Cout, int M, int N, int K) {
  const int lane = threadIdx.x & 31;
  const int wid  = threadIdx.x >> 5;
  const int wm   = wid & 3;       // 4 waves along M
  const int wn   = wid >> 2;      // 2 waves along N
  const int m0   = blockIdx.x * 128 + wm * 32;
  const int n0   = blockIdx.y * 128 + wn * 64;

  f32x8 acc[2][4] = {};
  for (int k0 = 0; k0 < K; k0 += 32) {
    if (k0 + 32 < K) {  // pull next-k fragment lines toward the WGP
      __builtin_prefetch(A + (size_t)(m0 + (lane & 15)) * K + k0 + 32, 0, 3);
      __builtin_prefetch(W + (size_t)(n0 + (lane & 15)) * K + k0 + 32, 0, 3);
    }
    bf16x16 af[2], bfv[4];
#pragma unroll
    for (int mi = 0; mi < 2; ++mi)
      af[mi] = load_a_frag(A, K, m0 + mi * 16, k0, lane);
#pragma unroll
    for (int ni = 0; ni < 4; ++ni)
      bfv[ni] = load_b_frag(W, K, n0 + ni * 16, k0, lane);
#pragma unroll
    for (int mi = 0; mi < 2; ++mi)
#pragma unroll
      for (int ni = 0; ni < 4; ++ni)
        acc[mi][ni] = wmma_bf16(af[mi], bfv[ni], acc[mi][ni]);
  }

  // C layout: lane col = lane&15, rows rb..rb+7 with rb = 8*(lane>>4)
  const int col = lane & 15;
  const int rb  = (lane >> 4) * 8;
#pragma unroll
  for (int mi = 0; mi < 2; ++mi)
#pragma unroll
    for (int ni = 0; ni < 4; ++ni)
#pragma unroll
      for (int r = 0; r < 8; ++r) {
        const int m = m0 + mi * 16 + rb + r;
        const int n = n0 + ni * 16 + col;
        const float v = acc[mi][ni][r];
        if (OUT_BF16) ((__bf16*)Cout)[(size_t)m * N + n] = (__bf16)v;
        else          ((float*)Cout)[(size_t)m * N + n]  = v;
      }
}

// ---------------------------------------------------------------------------
// RoPE + RMSNorm, in-place on the q/k slices of the qkv buffer.
// One wave per (token, head) 128-vector; lane handles 2 complex pairs.
// ---------------------------------------------------------------------------
__global__ void __launch_bounds__(256)
rope_rms_kernel(__bf16* __restrict__ qkv, const float* __restrict__ cosb,
                const float* __restrict__ sinb) {
  const int lane = threadIdx.x & 31;
  const int wid  = threadIdx.x >> 5;
  const int vec  = blockIdx.x * 8 + wid;      // (token, head) index
  const int total = ROWS_ * (NH_ + NKV_);
  if (vec >= total) return;
  const int token = vec / (NH_ + NKV_);
  const int hh    = vec % (NH_ + NKV_);       // 0..15 q heads, 16..17 k heads
  const int t     = token & (T_ - 1);
  const size_t cb = (hh < NH_) ? (size_t)hh * HD_
                               : (size_t)(NH_ * HD_) + (size_t)(hh - NH_) * HD_;
  __bf16* p = qkv + (size_t)token * LDQ_ + cb;

  const int i0 = lane * 2;                    // complex pair indices i0, i0+1
  const float x0 = (float)p[i0 * 2 + 0];
  const float x1 = (float)p[i0 * 2 + 1];
  const float x2 = (float)p[i0 * 2 + 2];
  const float x3 = (float)p[i0 * 2 + 3];
  const float c0 = cosb[(size_t)t * 64 + i0];
  const float s0 = sinb[(size_t)t * 64 + i0];
  const float c1 = cosb[(size_t)t * 64 + i0 + 1];
  const float s1 = sinb[(size_t)t * 64 + i0 + 1];
  const float r0 = x0 * c0 - x1 * s0;
  const float u0 = x0 * s0 + x1 * c0;
  const float r1 = x2 * c1 - x3 * s1;
  const float u1 = x2 * s1 + x3 * c1;

  float ss = r0 * r0 + u0 * u0 + r1 * r1 + u1 * u1;
#pragma unroll
  for (int m = 1; m <= 16; m <<= 1) ss += __shfl_xor(ss, m, 32);
  const float sc = rsqrtf(ss * (1.0f / 128.0f) + 1.1920929e-7f);

  p[i0 * 2 + 0] = (__bf16)(r0 * sc);
  p[i0 * 2 + 1] = (__bf16)(u0 * sc);
  p[i0 * 2 + 2] = (__bf16)(r1 * sc);
  p[i0 * 2 + 3] = (__bf16)(u1 * sc);
}

// ---------------------------------------------------------------------------
// Flash attention (causal, GQA). Block = (q-tile of 128 rows, head, batch).
// 8 waves; each wave owns 16 q rows. K tile (64x128) staged row-major in LDS
// via CDNA5 async global->LDS copies (ASYNCcnt), V tile transposed (128x64)
// in LDS, per-wave P (16x64) in LDS for the scores->A-fragment transpose.
// Online softmax in registers.
// ---------------------------------------------------------------------------
__global__ void __launch_bounds__(256)
attn_kernel(const __bf16* __restrict__ qkv, __bf16* __restrict__ yb) {
  __shared__ alignas(32) __bf16 sK[64][128];       // (s, d)
  __shared__ alignas(32) __bf16 sVt[128][72];      // (d, s) padded
  __shared__ alignas(32) __bf16 sP[8][16][64];     // per-wave probs (q, s)

  const int lane = threadIdx.x & 31;
  const int wid  = threadIdx.x >> 5;
  const int h    = blockIdx.y;
  const int b    = blockIdx.z;
  const int g    = h >> 3;                         // kv head
  const int q0   = blockIdx.x * 128;

  const __bf16* Qb = qkv + (size_t)b * T_ * LDQ_ + (size_t)h * HD_;
  const __bf16* Kb = qkv + (size_t)b * T_ * LDQ_ + (size_t)(NH_ * HD_) + (size_t)g * HD_;
  const __bf16* Vb = qkv + (size_t)b * T_ * LDQ_ + (size_t)(NH_ * HD_ + NKV_ * HD_) + (size_t)g * HD_;

  // Q fragments for this wave's 16 rows: resident all kernel (4 x k=32).
  bf16x16 qf[4];
#pragma unroll
  for (int c = 0; c < 4; ++c)
    qf[c] = load_a_frag(Qb, LDQ_, q0 + wid * 16, c * 32, lane);

  f32x8 o[8] = {};
  float mrow[8], lrow[8];
#pragma unroll
  for (int r = 0; r < 8; ++r) { mrow[r] = -3.0e38f; lrow[r] = 0.f; }

  const int col = lane & 15;
  const int rb  = (lane >> 4) * 8;
  const float sscale = 0.08838834764831845f;       // 1/sqrt(128)

  for (int s0 = 0; s0 < q0 + 128; s0 += 64) {
    __syncthreads();
    // Stage K tile (64x128): async 16B copies straight into LDS (no VGPRs).
#pragma unroll
    for (int i = threadIdx.x; i < 64 * 16; i += 256) {
      const int sr = i >> 4;
      const int dc = (i & 15) << 3;
      async_g2lds_b128(&sK[sr][dc], Kb + (size_t)(s0 + sr) * LDQ_ + dc);
    }
    // Stage V tile transposed (d-major) for contiguous B-fragment reads.
    for (int i = threadIdx.x; i < 64 * 64; i += 256) {
      const int sr = i >> 6;
      const int dp = (i & 63) << 1;
      const __bf16* vp = Vb + (size_t)(s0 + sr) * LDQ_ + dp;
      sVt[dp + 0][sr] = vp[0];
      sVt[dp + 1][sr] = vp[1];
    }
    wait_asynccnt0();          // this wave's async K copies landed in LDS
    __syncthreads();           // all waves' K/V staging visible

    // Scores: S(16 x 64) = Q(16x128) * K^T, 16 WMMAs.
    f32x8 sc[4] = {};
#pragma unroll
    for (int ni = 0; ni < 4; ++ni)
#pragma unroll
      for (int c = 0; c < 4; ++c) {
        bf16x16 kf = *(const bf16x16*)&sK[ni * 16 + col][c * 32 + ((lane >> 4) << 4)];
        sc[ni] = wmma_bf16(qf[c], kf, sc[ni]);
      }

    // Scale + causal mask + row max.
    float tmax[8];
#pragma unroll
    for (int r = 0; r < 8; ++r) tmax[r] = -3.0e38f;
#pragma unroll
    for (int ni = 0; ni < 4; ++ni) {
      const int sg = s0 + ni * 16 + col;
      const int qg = q0 + wid * 16 + rb;
#pragma unroll
      for (int r = 0; r < 8; ++r) {
        float v = sc[ni][r] * sscale;
        v = (sg <= qg + r) ? v : -3.0e38f;
        sc[ni][r] = v;
        tmax[r] = fmaxf(tmax[r], v);
      }
    }
#pragma unroll
    for (int r = 0; r < 8; ++r)
#pragma unroll
      for (int m = 1; m <= 8; m <<= 1)     // reduce across the 16 cols
        tmax[r] = fmaxf(tmax[r], __shfl_xor(tmax[r], m, 32));

    float alpha[8], rsum[8];
#pragma unroll
    for (int r = 0; r < 8; ++r) {
      const float mnew = fmaxf(mrow[r], tmax[r]);
      alpha[r] = __expf(mrow[r] - mnew);
      mrow[r]  = mnew;
      rsum[r]  = 0.f;
    }
    // P = exp(S - m), spill to per-wave LDS for the transpose to A-layout.
#pragma unroll
    for (int ni = 0; ni < 4; ++ni)
#pragma unroll
      for (int r = 0; r < 8; ++r) {
        const float p = __expf(sc[ni][r] - mrow[r]);
        rsum[r] += p;
        sP[wid][rb + r][ni * 16 + col] = (__bf16)p;
      }
#pragma unroll
    for (int r = 0; r < 8; ++r) {
#pragma unroll
      for (int m = 1; m <= 8; m <<= 1) rsum[r] += __shfl_xor(rsum[r], m, 32);
      lrow[r] = lrow[r] * alpha[r] + rsum[r];
    }
    // Rescale O, then O += P * V   (2 k-chunks x 8 d-tiles = 16 WMMAs).
#pragma unroll
    for (int ot = 0; ot < 8; ++ot)
#pragma unroll
      for (int r = 0; r < 8; ++r) o[ot][r] *= alpha[r];

#pragma unroll
    for (int c = 0; c < 2; ++c) {
      const bf16x16 pf = load_a_frag(&sP[wid][0][0], 64, 0, c * 32, lane);
#pragma unroll
      for (int ot = 0; ot < 8; ++ot) {
        bf16x16 vf = *(const bf16x16*)&sVt[ot * 16 + col][c * 32 + ((lane >> 4) << 4)];
        o[ot] = wmma_bf16(pf, vf, o[ot]);
      }
    }
  }

  // Epilogue: O / l -> yb[token, h*128 + d]  (bf16 for the output GEMM).
#pragma unroll
  for (int ot = 0; ot < 8; ++ot)
#pragma unroll
    for (int r = 0; r < 8; ++r) {
      const int tok = q0 + wid * 16 + rb + r;
      const float val = o[ot][r] / lrow[r];
      yb[((size_t)b * T_ + tok) * (size_t)HID_ + h * HD_ + ot * 16 + col] = (__bf16)val;
    }
}

// ---------------------------------------------------------------------------
// Host launcher
// ---------------------------------------------------------------------------
extern "C" void kernel_launch(void* const* d_in, const int* in_sizes, int n_in,
                              void* d_out, int out_size, void* d_ws, size_t ws_size,
                              hipStream_t stream) {
  (void)in_sizes; (void)n_in; (void)out_size; (void)ws_size;
  const float* x    = (const float*)d_in[0];
  const float* cosb = (const float*)d_in[1];
  const float* sinb = (const float*)d_in[2];
  const float* Wq   = (const float*)d_in[3];
  const float* Wk   = (const float*)d_in[4];
  const float* Wv   = (const float*)d_in[5];
  const float* Wo   = (const float*)d_in[6];

  char* ws = (char*)d_ws;
  __bf16* xb   = (__bf16*)(ws + 0);            // [8192, 2048]
  __bf16* Wcat = (__bf16*)(ws + 33554432ull);  // [2560, 2048]  Wq|Wk|Wv rows
  __bf16* Wob  = (__bf16*)(ws + 44040192ull);  // [2048, 2048]
  __bf16* qkv  = (__bf16*)(ws + 52428800ull);  // [8192, 2560]
  __bf16* yb   = (__bf16*)(ws + 94371840ull);  // [8192, 2048]

  auto cvt = [&](const float* src, __bf16* dst, long n) {
    const int blocks = (int)((n + 1023) / 1024);
    cvt_f32_bf16_kernel<<<dim3(blocks), dim3(256), 0, stream>>>(src, dst, n);
  };
  cvt(x,  xb,                                        (long)ROWS_ * HID_);
  cvt(Wq, Wcat,                                      (long)NH_ * HD_ * HID_);
  cvt(Wk, Wcat + (size_t)NH_ * HD_ * HID_,           (long)NKV_ * HD_ * HID_);
  cvt(Wv, Wcat + (size_t)(NH_ + NKV_) * HD_ * HID_,  (long)NKV_ * HD_ * HID_);
  cvt(Wo, Wob,                                       (long)HID_ * HID_);

  // qkv = xb * Wcat^T   (bf16 out)
  gemm_bf16_kernel<true><<<dim3(ROWS_ / 128, LDQ_ / 128), dim3(256), 0, stream>>>(
      xb, Wcat, qkv, ROWS_, LDQ_, HID_);

  // RoPE + RMSNorm in-place on q,k slices.
  rope_rms_kernel<<<dim3(ROWS_ * (NH_ + NKV_) / 8), dim3(256), 0, stream>>>(
      qkv, cosb, sinb);

  // Causal GQA flash attention -> yb (bf16).
  attn_kernel<<<dim3(T_ / 128, NH_, B_), dim3(256), 0, stream>>>(qkv, yb);

  // out = yb * Wo^T  (f32 out)
  gemm_bf16_kernel<false><<<dim3(ROWS_ / 128, HID_ / 128), dim3(256), 0, stream>>>(
      yb, Wob, d_out, ROWS_, HID_, HID_);
}
